// SetAbstraction_76544907149498
// MI455X (gfx1250) — matrix operand order repaired and checked by reading, not compiled, not run on hardware
//
#include <hip/hip_runtime.h>
#include <hip/hip_bf16.h>

// PointNet++ set-abstraction for MI455X (gfx1250, wave32).
// Stages: FPS -> gather set_loc -> kNN(top-32) -> gather+MLP(WMMA f16)+maxpool.

#define B_ 16
#define C_ 64
#define N_ 4096
#define D_ 3
#define M_ 1024
#define K_ 32

typedef __attribute__((ext_vector_type(16))) _Float16 v16h;
typedef __attribute__((ext_vector_type(8)))  float    v8f;

// ---------------------------------------------------------------------------
// Kernel 1: farthest point sampling. One block (1024 threads) per batch.
// ---------------------------------------------------------------------------
__global__ __launch_bounds__(1024) void fps_kernel(const float* __restrict__ loc,
                                                   int* __restrict__ set_idx) {
  const int b    = blockIdx.x;
  const int tid  = threadIdx.x;
  const int lane = tid & 31;
  const int wave = tid >> 5;

  __shared__ float sx[N_], sy[N_], sz[N_];
  __shared__ float red_v[32];
  __shared__ int   red_i[32];
  __shared__ int   s_last;

  const float* lb = loc + (size_t)b * 3 * N_;
  for (int i = tid; i < N_; i += 1024) {
    sx[i] = lb[i];
    sy[i] = lb[N_ + i];
    sz[i] = lb[2 * N_ + i];
  }
  __syncthreads();

  float dist[4] = {3.4e38f, 3.4e38f, 3.4e38f, 3.4e38f};
  int last = 0;

  for (int m = 0; m < M_; ++m) {
    const float lx = sx[last], ly = sy[last], lz = sz[last];
    float bestv = -1.0f;
    int   besti = 0;
#pragma unroll
    for (int i = 0; i < 4; ++i) {
      const int p = tid + i * 1024;
      const float dx = sx[p] - lx, dy = sy[p] - ly, dz = sz[p] - lz;
      const float d2 = dx * dx + dy * dy + dz * dz;
      dist[i] = fminf(dist[i], d2);
      if (dist[i] > bestv) { bestv = dist[i]; besti = p; }
    }
    // wave argmax, tie -> lower index
#pragma unroll
    for (int off = 16; off >= 1; off >>= 1) {
      const float ov = __shfl_xor(bestv, off);
      const int   oi = __shfl_xor(besti, off);
      if (ov > bestv || (ov == bestv && oi < besti)) { bestv = ov; besti = oi; }
    }
    if (lane == 0) { red_v[wave] = bestv; red_i[wave] = besti; }
    if (tid == 0) set_idx[b * M_ + m] = last;  // pre-update index (scan carry)
    __syncthreads();
    if (wave == 0) {
      float v = red_v[lane];
      int   i = red_i[lane];
#pragma unroll
      for (int off = 16; off >= 1; off >>= 1) {
        const float ov = __shfl_xor(v, off);
        const int   oi = __shfl_xor(i, off);
        if (ov > v || (ov == v && oi < i)) { v = ov; i = oi; }
      }
      if (lane == 0) s_last = i;
    }
    __syncthreads();
    last = s_last;
  }
}

// ---------------------------------------------------------------------------
// Kernel 2: gather set_loc = loc[:, :, set_idx]. One block per batch.
// ---------------------------------------------------------------------------
__global__ __launch_bounds__(1024) void gather_setloc_kernel(
    const float* __restrict__ loc, const int* __restrict__ set_idx,
    float* __restrict__ set_loc) {
  const int b = blockIdx.x;
  const int m = threadIdx.x;
  const int idx = set_idx[b * M_ + m];
#pragma unroll
  for (int d = 0; d < 3; ++d)
    set_loc[(size_t)b * 3 * M_ + d * M_ + m] = loc[(size_t)b * 3 * N_ + d * N_ + idx];
}

// ---------------------------------------------------------------------------
// Kernel 3: kNN (32 smallest d2 of 4096). One wave per query, 8 waves/block.
// ---------------------------------------------------------------------------
__global__ __launch_bounds__(256) void knn_kernel(const float* __restrict__ loc,
                                                  const float* __restrict__ set_loc,
                                                  int* __restrict__ nn_idx) {
  const int lane = threadIdx.x & 31;
  const int wave = threadIdx.x >> 5;
  const int q    = blockIdx.x * 8 + wave;
  const int b    = q >> 10;       // q / M_
  const int m    = q & (M_ - 1);  // q % M_

  const float* sl = set_loc + (size_t)b * 3 * M_;
  const float qx = sl[m], qy = sl[M_ + m], qz = sl[2 * M_ + m];

  const float* lx = loc + (size_t)b * 3 * N_;
  const float* ly = lx + N_;
  const float* lz = lx + 2 * N_;

  unsigned claimed[4] = {0u, 0u, 0u, 0u};

  // initial local scan: lane owns points p = j*32 + lane, j in [0,128)
  float bestd = 3.4e38f;
  int   bestj = 0;
  for (int j = 0; j < 128; ++j) {
    const int p = j * 32 + lane;
    const float dx = lx[p] - qx, dy = ly[p] - qy, dz = lz[p] - qz;
    const float d2 = dx * dx + dy * dy + dz * dz;
    if (d2 < bestd) { bestd = d2; bestj = j; }
  }

  int myidx = 0;
  for (int i = 0; i < K_; ++i) {
    float v  = bestd;
    int   pj = bestj * 32 + lane;
#pragma unroll
    for (int off = 16; off >= 1; off >>= 1) {
      const float ov = __shfl_xor(v, off);
      const int   op = __shfl_xor(pj, off);
      if (ov < v || (ov == v && op < pj)) { v = ov; pj = op; }
    }
    // all lanes now hold the global winner (v, pj)
    if ((pj & 31) == lane) {
      const int wj = pj >> 5;
      claimed[wj >> 5] |= 1u << (wj & 31);
      bestd = 3.4e38f;
      bestj = 0;
      for (int j = 0; j < 128; ++j) {
        if ((claimed[j >> 5] >> (j & 31)) & 1u) continue;
        const int p = j * 32 + lane;
        const float dx = lx[p] - qx, dy = ly[p] - qy, dz = lz[p] - qz;
        const float d2 = dx * dx + dy * dy + dz * dz;
        if (d2 < bestd) { bestd = d2; bestj = j; }
      }
    }
    if (lane == i) myidx = pj;
  }
  nn_idx[((size_t)b * M_ + m) * K_ + lane] = myidx;
}

// ---------------------------------------------------------------------------
// Kernel 4: gather + 3-layer pointnet MLP via WMMA f16 + k-maxpool.
// One wave per sampled point m; 8 waves (256 threads) per block.
// ---------------------------------------------------------------------------
__device__ inline v8f wmma_f16(v16h a, v16h b, v8f c) {
  return __builtin_amdgcn_wmma_f32_16x16x32_f16(false, a, false, b, (short)0, c,
                                                false, false);
}

// C-layout (row=outch) -> B-layout (K=inch) relayout via half-wave exchange.
// loT = acc rowTile (2c), hiT = acc rowTile (2c+1).
__device__ inline v16h relayout_b(v8f loT, v8f hiT, int hi) {
  v16h o;
#pragma unroll
  for (int v = 0; v < 8; ++v) {
    const float s0 = loT[v];
    const float s1 = hiT[v];
    const float x0 = __shfl_xor(s0, 16);
    const float x1 = __shfl_xor(s1, 16);
    o[v]     = (_Float16)(hi ? x1 : s0);
    o[8 + v] = (_Float16)(hi ? s1 : x0);
  }
  return o;
}

__global__ __launch_bounds__(256) void mlp_kernel(
    const float* __restrict__ feat, const float* __restrict__ loc,
    const float* __restrict__ set_loc, const int* __restrict__ nn_idx,
    const float* __restrict__ W1, const float* __restrict__ b1,
    const float* __restrict__ W2, const float* __restrict__ b2,
    const float* __restrict__ W3, const float* __restrict__ b3,
    float* __restrict__ set_feat) {
  const int tid  = threadIdx.x;
  const int lane = tid & 31;
  const int hi   = lane >> 4;
  const int ln   = lane & 15;

  // Weights packed in fragment order: offset = ((frag*32 + lane)*16 + half).
  __shared__ alignas(32) _Float16 w1f[12 * 512];  // 64x96 (padded from 67)
  __shared__ alignas(32) _Float16 w2f[8 * 512];   // 64x64
  __shared__ alignas(32) _Float16 w3f[16 * 512];  // 128x64
  __shared__ float sb1[64], sb2[64], sb3[128];

  for (int idx = tid; idx < 12 * 512; idx += 256) {
    const int h = idx & 15, l = (idx >> 4) & 31, f = idx >> 9;
    const int r = f / 3, c = f % 3;
    const int row = r * 16 + (l & 15);
    const int kk  = c * 32 + h + ((l >> 4) ? 16 : 0);
    w1f[idx] = (_Float16)((kk < 67) ? W1[row * 67 + kk] : 0.0f);
  }
  for (int idx = tid; idx < 8 * 512; idx += 256) {
    const int h = idx & 15, l = (idx >> 4) & 31, f = idx >> 9;
    const int r = f >> 1, c = f & 1;
    const int row = r * 16 + (l & 15);
    const int kk  = c * 32 + h + ((l >> 4) ? 16 : 0);
    w2f[idx] = (_Float16)W2[row * 64 + kk];
  }
  for (int idx = tid; idx < 16 * 512; idx += 256) {
    const int h = idx & 15, l = (idx >> 4) & 31, f = idx >> 9;
    const int r = f >> 1, c = f & 1;
    const int row = r * 16 + (l & 15);
    const int kk  = c * 32 + h + ((l >> 4) ? 16 : 0);
    w3f[idx] = (_Float16)W3[row * 64 + kk];
  }
  if (tid < 64) { sb1[tid] = b1[tid]; sb2[tid] = b2[tid]; }
  if (tid < 128) sb3[tid] = b3[tid];
  __syncthreads();

  const int q = blockIdx.x * 8 + (tid >> 5);
  const int b = q >> 10;
  const int m = q & (M_ - 1);

  const int* nn = nn_idx + ((size_t)b * M_ + m) * K_;
  const int p0 = nn[ln];        // column ln        (col-tile 0)
  const int p1 = nn[16 + ln];   // column 16 + ln   (col-tile 1)

  const float* F = feat + (size_t)b * C_ * N_;
  const float* L = loc + (size_t)b * 3 * N_;
  const float* SL = set_loc + (size_t)b * 3 * M_;
  float sl[3];
#pragma unroll
  for (int d = 0; d < 3; ++d) sl[d] = SL[d * M_ + m];

  const v8f vzero = {0.f, 0.f, 0.f, 0.f, 0.f, 0.f, 0.f, 0.f};

  // ---- Layer 1: 67 -> 64 -------------------------------------------------
  v8f acc1[4][2];
#pragma unroll
  for (int r = 0; r < 4; ++r) { acc1[r][0] = vzero; acc1[r][1] = vzero; }

#pragma unroll
  for (int c = 0; c < 3; ++c) {
    v16h bf[2];
#pragma unroll
    for (int t = 0; t < 2; ++t) {
      const int p = t ? p1 : p0;
#pragma unroll
      for (int h = 0; h < 16; ++h) {
        const int ch = c * 32 + h + (hi ? 16 : 0);  // B K-index (sequential)
        float v;
        if (ch < 64)      v = F[ch * N_ + p];
        else if (ch < 67) v = L[(ch - 64) * N_ + p] - sl[ch - 64];
        else              v = 0.0f;
        bf[t][h] = (_Float16)v;
      }
    }
#pragma unroll
    for (int r = 0; r < 4; ++r) {
      const v16h a = *(const v16h*)(w1f + ((r * 3 + c) * 32 + lane) * 16);
      acc1[r][0] = wmma_f16(a, bf[0], acc1[r][0]);
      acc1[r][1] = wmma_f16(a, bf[1], acc1[r][1]);
    }
  }
  // bias + relu
#pragma unroll
  for (int r = 0; r < 4; ++r)
#pragma unroll
    for (int t = 0; t < 2; ++t)
#pragma unroll
      for (int v = 0; v < 8; ++v)
        acc1[r][t][v] = fmaxf(acc1[r][t][v] + sb1[r * 16 + hi * 8 + v], 0.0f);

  // ---- Layer 2: 64 -> 64 -------------------------------------------------
  v16h bf2[2][2];
#pragma unroll
  for (int c = 0; c < 2; ++c)
#pragma unroll
    for (int t = 0; t < 2; ++t)
      bf2[c][t] = relayout_b(acc1[2 * c][t], acc1[2 * c + 1][t], hi);

  v8f acc2[4][2];
#pragma unroll
  for (int r = 0; r < 4; ++r) { acc2[r][0] = vzero; acc2[r][1] = vzero; }
#pragma unroll
  for (int c = 0; c < 2; ++c)
#pragma unroll
    for (int r = 0; r < 4; ++r) {
      const v16h a = *(const v16h*)(w2f + ((r * 2 + c) * 32 + lane) * 16);
      acc2[r][0] = wmma_f16(a, bf2[c][0], acc2[r][0]);
      acc2[r][1] = wmma_f16(a, bf2[c][1], acc2[r][1]);
    }
#pragma unroll
  for (int r = 0; r < 4; ++r)
#pragma unroll
    for (int t = 0; t < 2; ++t)
#pragma unroll
      for (int v = 0; v < 8; ++v)
        acc2[r][t][v] = fmaxf(acc2[r][t][v] + sb2[r * 16 + hi * 8 + v], 0.0f);

  // ---- Layer 3: 64 -> 128 ------------------------------------------------
  v16h bf3[2][2];
#pragma unroll
  for (int c = 0; c < 2; ++c)
#pragma unroll
    for (int t = 0; t < 2; ++t)
      bf3[c][t] = relayout_b(acc2[2 * c][t], acc2[2 * c + 1][t], hi);

  v8f acc3[8][2];
#pragma unroll
  for (int r = 0; r < 8; ++r) { acc3[r][0] = vzero; acc3[r][1] = vzero; }
#pragma unroll
  for (int c = 0; c < 2; ++c)
#pragma unroll
    for (int r = 0; r < 8; ++r) {
      const v16h a = *(const v16h*)(w3f + ((r * 2 + c) * 32 + lane) * 16);
      acc3[r][0] = wmma_f16(a, bf3[c][0], acc3[r][0]);
      acc3[r][1] = wmma_f16(a, bf3[c][1], acc3[r][1]);
    }

  // ---- bias + relu + maxpool over k=32 columns, write (b, 128, m) --------
#pragma unroll
  for (int r = 0; r < 8; ++r)
#pragma unroll
    for (int v = 0; v < 8; ++v) {
      const int row = r * 16 + hi * 8 + v;
      const float bia = sb3[row];
      float val = fmaxf(fmaxf(acc3[r][0][v] + bia, 0.0f),
                        fmaxf(acc3[r][1][v] + bia, 0.0f));
#pragma unroll
      for (int off = 8; off >= 1; off >>= 1)
        val = fmaxf(val, __shfl_xor(val, off));
      if (ln == 0)
        set_feat[(size_t)b * 128 * M_ + row * M_ + m] = val;
    }
}

// ---------------------------------------------------------------------------
extern "C" void kernel_launch(void* const* d_in, const int* in_sizes, int n_in,
                              void* d_out, int out_size, void* d_ws, size_t ws_size,
                              hipStream_t stream) {
  const float* feat = (const float*)d_in[0];
  const float* loc  = (const float*)d_in[1];
  const float* W1   = (const float*)d_in[2];
  const float* b1   = (const float*)d_in[3];
  const float* W2   = (const float*)d_in[4];
  const float* b2   = (const float*)d_in[5];
  const float* W3   = (const float*)d_in[6];
  const float* b3   = (const float*)d_in[7];

  float* out      = (float*)d_out;
  float* set_feat = out;                          // (16, 128, 1024)
  float* set_loc  = out + (size_t)B_ * 128 * M_;  // (16, 3, 1024)

  int* set_idx = (int*)d_ws;                          // 16K ints
  int* nn_idx  = (int*)((char*)d_ws + 65536);         // 512K ints

  fps_kernel<<<B_, 1024, 0, stream>>>(loc, set_idx);
  gather_setloc_kernel<<<B_, 1024, 0, stream>>>(loc, set_idx, set_loc);
  knn_kernel<<<(B_ * M_) / 8, 256, 0, stream>>>(loc, set_loc, nn_idx);
  mlp_kernel<<<(B_ * M_) / 8, 256, 0, stream>>>(feat, loc, set_loc, nn_idx,
                                                W1, b1, W2, b2, W3, b3, set_feat);
}
